// MultiHeadAttention_44057774523228
// MI455X (gfx1250) — matrix operand order repaired and checked by reading, not compiled, not run on hardware
//
#include <hip/hip_runtime.h>

typedef _Float16 v16h __attribute__((ext_vector_type(16)));
typedef _Float16 v8h  __attribute__((ext_vector_type(8)));
typedef float    v8f  __attribute__((ext_vector_type(8)));
typedef int      gv4i __attribute__((vector_size(16)));   // int4, GCC-style

#define HIDDEN   1024
#define HEADS    16
#define HEAD_DIM 64
#define BATCH    2
#define SEQ      2048
#define MROWS    (BATCH * SEQ)   // 4096

#define AS1 __attribute__((address_space(1)))
#define AS3 __attribute__((address_space(3)))

union AF { v16h v; v8h h2[2]; };

__device__ __forceinline__ v8f wmma_f16(v16h a, v16h b, v8f c) {
  // (neg_a, A, neg_b, B, c_mod, C, reuse_a, reuse_b)
  return __builtin_amdgcn_wmma_f32_16x16x32_f16(false, a, false, b, (short)0, c,
                                                false, false);
}

// Async global -> LDS b128 copy (ASYNCcnt-tracked), builtin with asm fallback.
__device__ __forceinline__ void async_copy_b128(const void* g, void* l) {
#if __has_builtin(__builtin_amdgcn_global_load_async_to_lds_b128)
  __builtin_amdgcn_global_load_async_to_lds_b128(
      (AS1 gv4i*)(uintptr_t)g, (AS3 gv4i*)l, 0, 0);
#else
  unsigned int loff = (unsigned int)(uintptr_t)(AS3 void*)l;
  unsigned long long ga = (unsigned long long)(uintptr_t)g;
  asm volatile("global_load_async_to_lds_b128 %0, %1, off"
               :: "v"(loff), "v"(ga) : "memory");
#endif
}

__device__ __forceinline__ void wait_async0() {
#if __has_builtin(__builtin_amdgcn_s_wait_asynccnt)
  __builtin_amdgcn_s_wait_asynccnt(0);
#else
  asm volatile("s_wait_asynccnt 0x0" ::: "memory");
#endif
}

// ---------------------------------------------------------------------------
// f32 -> f16 conversion (grid-stride)
// ---------------------------------------------------------------------------
__global__ void cvt_f32_f16(const float* __restrict__ src,
                            _Float16* __restrict__ dst, int n) {
  int i = blockIdx.x * blockDim.x + threadIdx.x;
  int stride = gridDim.x * blockDim.x;
  for (; i < n; i += stride) dst[i] = (_Float16)src[i];
}

// ---------------------------------------------------------------------------
// Y[M,N] = X[M,K] @ W[N,K]^T + bias   (f16 in, f32 accum, f16 or f32 out)
// Block: 128 threads = 4 waves (2x2), each wave owns a 64(M)x32(N) tile ->
// 128x64 block tile. X (128x64) / W (64x64) panels staged in LDS via async
// b128 copies; each 64-deep K stage runs 2 x (4x2) WMMAs per wave from LDS.
// Register budget: 8 acc (64 VGPR) + 4 A-frag (32) + 2 B-frag (16) -> no spill.
// A-frag: lane r holds row r; halfs at k = {half*8..+7} and {16+half*8..+7}.
// B-frag: lane r holds col r (= row r of W); halfs at k = {half*8..+7} and {16+half*8..+7}.
// D:      VGPR i -> row i + 8*half, col = lane%16.
// ---------------------------------------------------------------------------
__global__ void __launch_bounds__(128, 1)
gemm_wmma_kernel(const _Float16* __restrict__ X,
                 const _Float16* __restrict__ W,
                 const float* __restrict__ bias,
                 _Float16* __restrict__ Yh,
                 float* __restrict__ Yf,
                 int M, int N, int K) {
  const int t    = threadIdx.x;          // 0..127
  const int lane = t & 31;
  const int wave = t >> 5;
  const int r    = lane & 15;
  const int hf   = lane >> 4;

  const int mblk = blockIdx.y * 128;
  const int nblk = blockIdx.x * 64;
  const int m0   = mblk + (wave >> 1) * 64;
  const int n0   = nblk + (wave & 1) * 32;

  __shared__ __align__(16) _Float16 xs[128][64];   // X rows   [row][k]  16 KB
  __shared__ __align__(16) _Float16 wsm[64][64];   // W rows   [col][k]   8 KB

  v8f acc[4][2] = {};

  const _Float16* gx = X + (size_t)(mblk + t) * K;           // stage X row t
  const _Float16* gw = W + (size_t)(nblk + (t >> 1)) * K;    // stage W row t/2
  const int whalf = (t & 1) * 32;                            // 32-half column

  for (int k0 = 0; k0 < K; k0 += 64) {
    // ---- async stage: X 128x64 (8 b128/thread), W 64x64 (4 b128/thread) ----
#pragma unroll
    for (int j = 0; j < 8; ++j)
      async_copy_b128(gx + k0 + j * 8, &xs[t][j * 8]);
#pragma unroll
    for (int j = 0; j < 4; ++j)
      async_copy_b128(gw + k0 + whalf + j * 8, &wsm[t >> 1][whalf + j * 8]);
    __builtin_prefetch(gx + k0 + 64, 0, 1);   // global_prefetch_b8 (next stage)
    __builtin_prefetch(gw + k0 + 64, 0, 1);
    wait_async0();
    __syncthreads();

    // ---- compute: 2 K-substeps x (4x2) WMMAs -------------------------------
#pragma unroll
    for (int ks = 0; ks < 2; ++ks) {
      AF bf[2];
#pragma unroll
      for (int sn = 0; sn < 2; ++sn) {
        const _Float16* rp = &wsm[(wave & 1) * 32 + sn * 16 + r][ks * 32];
        bf[sn].h2[0] = *(const v8h*)(rp + hf * 8);
        bf[sn].h2[1] = *(const v8h*)(rp + 16 + hf * 8);
      }
#pragma unroll
      for (int sm = 0; sm < 4; ++sm) {
        const _Float16* rp = &xs[(wave >> 1) * 64 + sm * 16 + r][ks * 32];
        AF a;
        a.h2[0] = *(const v8h*)(rp + hf * 8);
        a.h2[1] = *(const v8h*)(rp + 16 + hf * 8);
        acc[sm][0] = wmma_f16(a.v, bf[0].v, acc[sm][0]);
        acc[sm][1] = wmma_f16(a.v, bf[1].v, acc[sm][1]);
      }
    }
    __syncthreads();   // protect LDS before next stage's async writes
  }

#pragma unroll
  for (int sm = 0; sm < 4; ++sm)
#pragma unroll
    for (int sn = 0; sn < 2; ++sn) {
      const int col = n0 + sn * 16 + r;
      const float bv = bias ? bias[col] : 0.0f;
#pragma unroll
      for (int i = 0; i < 8; ++i) {
        const int row = m0 + sm * 16 + hf * 8 + i;
        const float y = acc[sm][sn][i] + bv;
        if (Yf) Yf[(size_t)row * N + col] = y;
        else    Yh[(size_t)row * N + col] = (_Float16)y;
      }
    }
}

// ---------------------------------------------------------------------------
// Flash attention: one wave per (batch, head, 16-row q-tile).
// Streams over key blocks of 32; scores + P@V on WMMA, softmax in f32 VALU.
// ---------------------------------------------------------------------------
__global__ void attn_kernel(const _Float16* __restrict__ Q,
                            const _Float16* __restrict__ Kp,
                            const _Float16* __restrict__ V,
                            _Float16* __restrict__ O) {
  const int lane = threadIdx.x & 31;
  const int wave = threadIdx.x >> 5;
  const int r    = lane & 15;
  const int hf   = lane >> 4;

  const int task = blockIdx.x * 4 + wave;      // 0 .. 4095
  const int qt = task & 127;                   // q tile (S/16 = 128)
  const int h  = (task >> 7) & (HEADS - 1);
  const int b  = task >> 11;

  // per-wave LDS: vt = V tile transposed [64 d][32 key], pm = P [16 q][32 key]
  __shared__ __align__(32) _Float16 lds[4][64 * 32 + 16 * 32];
  _Float16* vt = lds[wave];
  _Float16* pm = lds[wave] + 64 * 32;

  const int q0 = qt * 16;
  const size_t baseQ = ((size_t)(b * SEQ + q0 + r)) * HIDDEN + h * HEAD_DIM;

  AF aq[2];
#pragma unroll
  for (int c = 0; c < 2; ++c) {
    aq[c].h2[0] = *(const v8h*)(Q + baseQ + c * 32 + hf * 8);
    aq[c].h2[1] = *(const v8h*)(Q + baseQ + c * 32 + 16 + hf * 8);
  }

  v8f o[4] = {};
  float mrow[8], lrow[8];
#pragma unroll
  for (int i = 0; i < 8; ++i) { mrow[i] = -1e30f; lrow[i] = 0.0f; }

  const float sc = 0.125f;  // 1/sqrt(HEAD_DIM)

  for (int kb = 0; kb < SEQ; kb += 32) {
    // ---- scores: 16x32 block as two 16x16 WMMA tiles, K=64 in two chunks ---
    v8f s[2] = {};
#pragma unroll
    for (int ns = 0; ns < 2; ++ns) {
      const size_t baseK =
          ((size_t)(b * SEQ + kb + ns * 16 + r)) * HIDDEN + h * HEAD_DIM;
      AF b0, b1;
      b0.h2[0] = *(const v8h*)(Kp + baseK + hf * 8);
      b0.h2[1] = *(const v8h*)(Kp + baseK + 16 + hf * 8);
      b1.h2[0] = *(const v8h*)(Kp + baseK + 32 + hf * 8);
      b1.h2[1] = *(const v8h*)(Kp + baseK + 32 + 16 + hf * 8);
      s[ns] = wmma_f16(aq[0].v, b0.v, s[ns]);
      s[ns] = wmma_f16(aq[1].v, b1.v, s[ns]);
    }

    // ---- stage V tile transposed into LDS: vt[d][key] ----------------------
    {
      const size_t baseV = ((size_t)(b * SEQ + kb + lane)) * HIDDEN + h * HEAD_DIM;
#pragma unroll
      for (int d0 = 0; d0 < HEAD_DIM; d0 += 8) {
        v8h vv = *(const v8h*)(V + baseV + d0);
#pragma unroll
        for (int e = 0; e < 8; ++e) vt[(d0 + e) * 32 + lane] = vv[e];
      }
    }

    // ---- online softmax over the 16x32 block -------------------------------
#pragma unroll
    for (int i = 0; i < 8; ++i) {
      float v0 = s[0][i] * sc, v1 = s[1][i] * sc;
      float mx = fmaxf(v0, v1);
#pragma unroll
      for (int msk = 8; msk >= 1; msk >>= 1)
        mx = fmaxf(mx, __shfl_xor(mx, msk, 32));
      const float mnew = fmaxf(mrow[i], mx);
      const float rescale = __expf(mrow[i] - mnew);
      mrow[i] = mnew;
      lrow[i] *= rescale;
#pragma unroll
      for (int c = 0; c < 4; ++c) o[c][i] *= rescale;
      const float p0 = __expf(v0 - mnew);
      const float p1 = __expf(v1 - mnew);
      float ps = p0 + p1;
#pragma unroll
      for (int msk = 8; msk >= 1; msk >>= 1) ps += __shfl_xor(ps, msk, 32);
      lrow[i] += ps;
      const int row = hf * 8 + i;             // D-layout row of this element
      pm[row * 32 + r]      = (_Float16)p0;
      pm[row * 32 + 16 + r] = (_Float16)p1;
    }
    __syncthreads();  // uniform trip count across all 4 waves -> legal

    // ---- O += P @ V --------------------------------------------------------
    AF ap;
    ap.h2[0] = *(const v8h*)(pm + r * 32 + hf * 8);
    ap.h2[1] = *(const v8h*)(pm + r * 32 + 16 + hf * 8);
#pragma unroll
    for (int c = 0; c < 4; ++c) {
      AF bv;
      bv.h2[0] = *(const v8h*)(vt + (c * 16 + r) * 32 + hf * 8);
      bv.h2[1] = *(const v8h*)(vt + (c * 16 + r) * 32 + 16 + hf * 8);
      o[c] = wmma_f16(ap.v, bv.v, o[c]);
    }
    __syncthreads();
  }

  // ---- normalize and write [B,S,HIDDEN] f16 --------------------------------
#pragma unroll
  for (int i = 0; i < 8; ++i) {
    const float inv = 1.0f / lrow[i];
    const int row = q0 + hf * 8 + i;
    const size_t base = ((size_t)(b * SEQ + row)) * HIDDEN + h * HEAD_DIM;
#pragma unroll
    for (int c = 0; c < 4; ++c)
      O[base + c * 16 + r] = (_Float16)(o[c][i] * inv);
  }
}

// ---------------------------------------------------------------------------
extern "C" void kernel_launch(void* const* d_in, const int* in_sizes, int n_in,
                              void* d_out, int out_size, void* d_ws, size_t ws_size,
                              hipStream_t stream) {
  const float* x  = (const float*)d_in[0];
  const float* Wq = (const float*)d_in[1];
  const float* bq = (const float*)d_in[2];
  const float* Wk = (const float*)d_in[3];
  const float* bk = (const float*)d_in[4];
  const float* Wv = (const float*)d_in[5];
  const float* bv = (const float*)d_in[6];
  const float* Wo = (const float*)d_in[7];
  const float* bo = (const float*)d_in[8];
  float* out = (float*)d_out;

  const size_t XE = (size_t)MROWS * HIDDEN;    // 4 Mi elems
  const size_t WE = (size_t)HIDDEN * HIDDEN;   // 1 Mi elems

  char* p = (char*)d_ws;
  _Float16* xh  = (_Float16*)p; p += XE * 2;
  _Float16* wqh = (_Float16*)p; p += WE * 2;
  _Float16* wkh = (_Float16*)p; p += WE * 2;
  _Float16* wvh = (_Float16*)p; p += WE * 2;
  _Float16* woh = (_Float16*)p; p += WE * 2;
  _Float16* qh  = (_Float16*)p; p += XE * 2;
  _Float16* kh  = (_Float16*)p; p += XE * 2;
  _Float16* vh  = (_Float16*)p; p += XE * 2;
  _Float16* ah  = (_Float16*)p; p += XE * 2;   // total 48 MiB

  // f32 -> f16
  cvt_f32_f16<<<2048, 256, 0, stream>>>(x,  xh,  (int)XE);
  cvt_f32_f16<<<1024, 256, 0, stream>>>(Wq, wqh, (int)WE);
  cvt_f32_f16<<<1024, 256, 0, stream>>>(Wk, wkh, (int)WE);
  cvt_f32_f16<<<1024, 256, 0, stream>>>(Wv, wvh, (int)WE);
  cvt_f32_f16<<<1024, 256, 0, stream>>>(Wo, woh, (int)WE);

  // Q/K/V projections (f16 out): block tile 128x64, 4 waves of 64x32
  dim3 ggrid(HIDDEN / 64, MROWS / 128);  // (16, 32)
  gemm_wmma_kernel<<<ggrid, 128, 0, stream>>>(xh, wqh, bq, qh, nullptr,
                                              MROWS, HIDDEN, HIDDEN);
  gemm_wmma_kernel<<<ggrid, 128, 0, stream>>>(xh, wkh, bk, kh, nullptr,
                                              MROWS, HIDDEN, HIDDEN);
  gemm_wmma_kernel<<<ggrid, 128, 0, stream>>>(xh, wvh, bv, vh, nullptr,
                                              MROWS, HIDDEN, HIDDEN);

  // attention: B*H*(S/16) = 4096 wave-tasks, 4 waves/block
  attn_kernel<<<1024, 128, 0, stream>>>(qh, kh, vh, ah);

  // output projection (f32 out -> d_out)
  gemm_wmma_kernel<<<ggrid, 128, 0, stream>>>(ah, woh, bo, nullptr, out,
                                              MROWS, HIDDEN, HIDDEN);
}
